// MultiHeadSelfAttention_28930899706483
// MI455X (gfx1250) — matrix operand order repaired and checked by reading, compiled, not run on hardware
//
#include <hip/hip_runtime.h>

// ---------------------------------------------------------------------------
// Types for CDNA5 WMMA (wave32, 16x16x32 bf16 -> f32 accumulate)
// ---------------------------------------------------------------------------
typedef __attribute__((ext_vector_type(16))) __bf16 v16bf;
typedef __attribute__((ext_vector_type(8)))  float  v8f;
typedef __attribute__((ext_vector_type(4)))  unsigned int u32x4;
typedef __attribute__((ext_vector_type(4)))  float  f32x4;

static __device__ __forceinline__ unsigned short f2bf(float f) {
  union { float f; unsigned int u; } v; v.f = f;
  unsigned int u = v.u;
  unsigned int r = (u + 0x7FFFu + ((u >> 16) & 1u)) >> 16;  // RNE
  return (unsigned short)r;
}

static __device__ __forceinline__ void store_out(float* p, float v)          { *p = v; }
static __device__ __forceinline__ void store_out(unsigned short* p, float v) { *p = f2bf(v); }

// Load 16 contiguous elements as bf16 pairs packed into two 16B chunks.
static __device__ __forceinline__ void ld16(const float* p, u32x4& x, u32x4& y) {
  const f32x4* q = (const f32x4*)p;
  f32x4 a = q[0], b = q[1], c = q[2], d = q[3];
  x[0] = (unsigned)f2bf(a[0]) | ((unsigned)f2bf(a[1]) << 16);
  x[1] = (unsigned)f2bf(a[2]) | ((unsigned)f2bf(a[3]) << 16);
  x[2] = (unsigned)f2bf(b[0]) | ((unsigned)f2bf(b[1]) << 16);
  x[3] = (unsigned)f2bf(b[2]) | ((unsigned)f2bf(b[3]) << 16);
  y[0] = (unsigned)f2bf(c[0]) | ((unsigned)f2bf(c[1]) << 16);
  y[1] = (unsigned)f2bf(c[2]) | ((unsigned)f2bf(c[3]) << 16);
  y[2] = (unsigned)f2bf(d[0]) | ((unsigned)f2bf(d[1]) << 16);
  y[3] = (unsigned)f2bf(d[2]) | ((unsigned)f2bf(d[3]) << 16);
}
static __device__ __forceinline__ void ld16(const unsigned short* p, u32x4& x, u32x4& y) {
  x = ((const u32x4*)p)[0];
  y = ((const u32x4*)p)[1];
}

union Frag16 { u32x4 q[2]; v16bf v; };

// Load a 16-element bf16 fragment as two 16B chunks from LDS (byte offsets).
static __device__ __forceinline__ v16bf ld_frag2(const unsigned short* lds,
                                                 int byte0, int byte1) {
  Frag16 f;
  f.q[0] = *(const u32x4*)((const char*)lds + byte0);
  f.q[1] = *(const u32x4*)((const char*)lds + byte1);
  return f.v;
}

static __device__ __forceinline__ v8f wmma_bf16(v16bf a, v16bf b, v8f c) {
  // D = A(16x32 bf16) * B(32x16 bf16) + C(16x16 f32)
  return __builtin_amdgcn_wmma_f32_16x16x32_bf16(false, a, false, b,
                                                 (short)0, c, false, false);
}

// ---------------------------------------------------------------------------
// 16-lane butterfly reductions in pure DPP (no LDS traffic):
// xor1 (quad_perm[1,0,3,2]=0xB1), xor2 (quad_perm[2,3,0,1]=0x4E),
// 8-group mirror (row_half_mirror=0x141), 16-group mirror (row_mirror=0x140).
// Each 16-lane half of the wave reduces independently (matches C-layout rows).
// ---------------------------------------------------------------------------
template <int CTRL>
static __device__ __forceinline__ float dpp_movf(float x) {
  int xi = __builtin_bit_cast(int, x);
  int r  = __builtin_amdgcn_update_dpp(xi, xi, CTRL, 0xF, 0xF, true);
  return __builtin_bit_cast(float, r);
}
static __device__ __forceinline__ float red_max16(float x) {
  x = fmaxf(x, dpp_movf<0x0B1>(x));
  x = fmaxf(x, dpp_movf<0x04E>(x));
  x = fmaxf(x, dpp_movf<0x141>(x));
  x = fmaxf(x, dpp_movf<0x140>(x));
  return x;
}
static __device__ __forceinline__ float red_sum16(float x) {
  x += dpp_movf<0x0B1>(x);
  x += dpp_movf<0x04E>(x);
  x += dpp_movf<0x141>(x);
  x += dpp_movf<0x140>(x);
  return x;
}

// Workgroup-relative LDS byte offset (low 32 bits of the generic address).
static __device__ __forceinline__ unsigned lds_off(const void* p) {
  return (unsigned)(unsigned long long)p;
}

// ---------------------------------------------------------------------------
// Kernel 1/3: C[M,N] = A[M,K] @ W[K,N] + bias   (A fp32 or bf16, C fp32 or bf16)
// Block: 128 threads (4 waves). Block tile 64x64, K-step 32.
// ---------------------------------------------------------------------------
template <typename AT, typename OT>
__global__ __launch_bounds__(128) void gemm_bias_kernel(
    const AT* __restrict__ A, const float* __restrict__ W,
    const float* __restrict__ bias, OT* __restrict__ C,
    int M, int N, int K) {
  __shared__ __attribute__((aligned(16))) unsigned short lds_a[64 * 32]; // [row][k]
  __shared__ __attribute__((aligned(16))) unsigned short lds_w[64 * 32]; // [n][k] (n-major)

  const int tid  = threadIdx.x;
  const int wave = tid >> 5;
  const int lane = tid & 31;
  const int hl   = lane >> 4;
  const int ln   = lane & 15;
  const int m0   = blockIdx.y * 64;
  const int n0   = blockIdx.x * 64;

  v8f acc[4];
#pragma unroll
  for (int t = 0; t < 4; ++t) acc[t] = v8f{};

  const int ar = tid >> 1, ah = tid & 1;  // A loader: row, 16-elem half

  for (int k0 = 0; k0 < K; k0 += 32) {
    // A tile 64x32 -> bf16 LDS, b128 global loads + b128 LDS stores
    {
      u32x4 x, y;
      ld16(&A[(size_t)(m0 + ar) * K + k0 + ah * 16], x, y);
      *(u32x4*)&lds_a[ar * 32 + ah * 16]     = x;
      *(u32x4*)&lds_a[ar * 32 + ah * 16 + 8] = y;
    }
    // W tile 32x64 -> bf16 LDS stored n-major [n][k]; b128 loads, b16 stores
#pragma unroll
    for (int i = 0; i < 4; ++i) {
      int c  = i * 128 + tid;
      int kr = c >> 4, nq = c & 15;
      f32x4 w = *(const f32x4*)&W[(size_t)(k0 + kr) * N + n0 + nq * 4];
#pragma unroll
      for (int e = 0; e < 4; ++e)
        lds_w[(nq * 4 + e) * 32 + kr] = f2bf(w[e]);
    }
    // Prefetch next K slice into cache (global_prefetch_b8)
    if (k0 + 32 < K) {
      __builtin_prefetch(&A[(size_t)(m0 + ar) * K + k0 + 32 + ah * 16], 0, 3);
      __builtin_prefetch(&W[(size_t)(k0 + 32 + (tid & 31)) * N + n0 + (tid >> 5) * 16], 0, 3);
    }
    __syncthreads();

    // A fragment: lane row = ln; k chunk hl*8 and +16  (16-bit A layout)
    const int arow = wave * 16 + ln;
    const int aoff = arow * 64 + hl * 16;  // bytes (row stride 64B)
    v16bf af = ld_frag2(lds_a, aoff, aoff + 32);
#pragma unroll
    for (int t = 0; t < 4; ++t) {
      const int ncol = t * 16 + ln;
      const int boff = ncol * 64 + hl * 16;
      v16bf bf = ld_frag2(lds_w, boff, boff + 32);
      acc[t] = wmma_bf16(af, bf, acc[t]);
    }
    __syncthreads();
  }

  // Epilogue: C layout -> slot r is row (r + hl*8), col ln within each subtile
#pragma unroll
  for (int t = 0; t < 4; ++t) {
    const int nc = n0 + t * 16 + ln;
    const float bv = bias[nc];
#pragma unroll
    for (int r = 0; r < 8; ++r) {
      const int mr = m0 + wave * 16 + hl * 8 + r;
      store_out(&C[(size_t)mr * N + nc], acc[t][r] + bv);
    }
  }
}

// ---------------------------------------------------------------------------
// Async DMA of one 64x64 bf16 K tile + V tile into LDS (8 x b128 per thread).
// ---------------------------------------------------------------------------
static __device__ __forceinline__ void issue_tile_async(
    unsigned long long saddr, unsigned kbase, unsigned vsbase,
    int j, int tid, int hc) {
#pragma unroll
  for (int i = 0; i < 4; ++i) {
    int c   = i * 128 + tid;          // 16B chunk id, 0..511
    int row = c >> 3, cc = c & 7;
    unsigned lk = kbase + row * 128 + cc * 16;
    unsigned vk = (unsigned)(((j * 64 + row) * 3072 + 1024 + hc + cc * 8) * 2);
    asm volatile("global_load_async_to_lds_b128 %0, %1, %2"
                 :: "v"(lk), "v"(vk), "s"(saddr) : "memory");
    unsigned lv = vsbase + row * 128 + cc * 16;
    unsigned vv = (unsigned)(((j * 64 + row) * 3072 + 2048 + hc + cc * 8) * 2);
    asm volatile("global_load_async_to_lds_b128 %0, %1, %2"
                 :: "v"(lv), "v"(vv), "s"(saddr) : "memory");
  }
}

// ---------------------------------------------------------------------------
// Kernel 2/3: fused flash attention per (b,h), double-buffered async staging.
// qkv: [B*T, 3072] bf16; out: [B*T, 1024] bf16 (head-merged)
// ---------------------------------------------------------------------------
__global__ __launch_bounds__(128) void attn_kernel(
    const unsigned short* __restrict__ qkv, unsigned short* __restrict__ out) {
  __shared__ __attribute__((aligned(16))) unsigned short lds_q[64 * 64];      // [q][dk]
  __shared__ __attribute__((aligned(16))) unsigned short lds_k[2][64 * 64];   // [key][dk]
  __shared__ __attribute__((aligned(16))) unsigned short lds_vs[2][64 * 64];  // V staging [key][dk]
  __shared__ __attribute__((aligned(16))) unsigned short lds_v[64 * 64];      // [dk][key]
  __shared__ __attribute__((aligned(16))) unsigned short lds_p[4][16 * 64];   // per-wave P

  const int tid  = threadIdx.x;
  const int wave = tid >> 5;
  const int lane = tid & 31;
  const int hl   = lane >> 4;
  const int ln   = lane & 15;

  const int qt = blockIdx.x;          // query tile 0..31
  const int bh = blockIdx.y;          // 0..31
  const int b  = bh >> 4;
  const int h  = bh & 15;
  const size_t rowbase = (size_t)b * 2048;
  const int hc = h * 64;

  const unsigned long long saddr =
      (unsigned long long)qkv + (unsigned long long)rowbase * 3072ull * 2ull;
  const unsigned qb  = lds_off(&lds_q[0]);
  const unsigned kb0 = lds_off(&lds_k[0][0]), kb1 = lds_off(&lds_k[1][0]);
  const unsigned vb0 = lds_off(&lds_vs[0][0]), vb1 = lds_off(&lds_vs[1][0]);

  // Prologue: async-copy Q tile, then K/V tile 0.
#pragma unroll
  for (int i = 0; i < 4; ++i) {
    int c   = i * 128 + tid;
    int row = c >> 3, cc = c & 7;
    unsigned lq = qb + row * 128 + cc * 16;
    unsigned vq = (unsigned)(((qt * 64 + row) * 3072 + hc + cc * 8) * 2);
    asm volatile("global_load_async_to_lds_b128 %0, %1, %2"
                 :: "v"(lq), "v"(vq), "s"(saddr) : "memory");
  }
  issue_tile_async(saddr, kb0, vb0, 0, tid, hc);

  v8f o[4];
  float m[8], l[8];
#pragma unroll
  for (int t = 0; t < 4; ++t) o[t] = v8f{};
#pragma unroll
  for (int r = 0; r < 8; ++r) { m[r] = -3.0e38f; l[r] = 0.0f; }

  const int qrow = wave * 16 + ln;
  v16bf qa0 = {}, qa1 = {};           // loop-invariant Q fragments (dk 0..31, 32..63)

  for (int j = 0; j < 32; ++j) {
    const int buf = j & 1;
    // Drain tile j (and Q on the first pass); then make visible to all waves.
    asm volatile("s_wait_asynccnt 0" ::: "memory");
    __syncthreads();
    if (j == 0) {  // hoist Q fragments into registers once
      const int aq = qrow * 128 + hl * 16;
      qa0 = ld_frag2(lds_q, aq, aq + 32);
      qa1 = ld_frag2(lds_q, aq + 64, aq + 96);
    }
    // Overlap: start DMA of tile j+1 into the other buffer.
    if (j < 31)
      issue_tile_async(saddr, buf ? kb0 : kb1, buf ? vb0 : vb1, j + 1, tid, hc);

    // Transpose V staging [key][dk] -> lds_v [dk][key] (b128 reads, b16 writes)
    {
      const unsigned short* vs = &lds_vs[buf][0];
#pragma unroll
      for (int i = 0; i < 4; ++i) {
        int c = i * 128 + tid;
        int key = c >> 3, d8 = c & 7;
        u32x4 vv = *(const u32x4*)&vs[key * 64 + d8 * 8];
        const unsigned short* e = (const unsigned short*)&vv;
#pragma unroll
        for (int q = 0; q < 8; ++q)
          lds_v[(d8 * 8 + q) * 64 + key] = e[q];
      }
    }
    __syncthreads();

    // S = Q @ K^T  (per wave: 16 q-rows x 64 keys)
    const unsigned short* kt = &lds_k[buf][0];
    v8f s[4];
#pragma unroll
    for (int t = 0; t < 4; ++t) s[t] = v8f{};
#pragma unroll
    for (int ks = 0; ks < 2; ++ks) {
      v16bf af = ks ? qa1 : qa0;
#pragma unroll
      for (int t = 0; t < 4; ++t) {
        const int key  = t * 16 + ln;
        const int boff = key * 128 + ks * 64 + hl * 16;
        v16bf bf = ld_frag2(kt, boff, boff + 32);
        s[t] = wmma_bf16(af, bf, s[t]);
      }
    }

    // Online softmax; scale 1/sqrt(64)=0.125 folded in. DPP reductions.
#pragma unroll
    for (int r = 0; r < 8; ++r) {
#pragma unroll
      for (int t = 0; t < 4; ++t) s[t][r] *= 0.125f;
      float mx = fmaxf(fmaxf(s[0][r], s[1][r]), fmaxf(s[2][r], s[3][r]));
      mx = red_max16(mx);
      float mnew  = fmaxf(m[r], mx);
      float alpha = __expf(m[r] - mnew);
      m[r] = mnew;
      float rs = 0.0f;
#pragma unroll
      for (int t = 0; t < 4; ++t) {
        float p = __expf(s[t][r] - mnew);
        s[t][r] = p;
        rs += p;
      }
      rs = red_sum16(rs);
      l[r] = l[r] * alpha + rs;
#pragma unroll
      for (int t = 0; t < 4; ++t) o[t][r] *= alpha;
    }

    // Re-distribute P from C-layout to A-layout via per-wave LDS scratch.
    unsigned short* pw = &lds_p[wave][0];
#pragma unroll
    for (int t = 0; t < 4; ++t)
#pragma unroll
      for (int r = 0; r < 8; ++r)
        pw[(r + hl * 8) * 64 + t * 16 + ln] = f2bf(s[t][r]);
    __builtin_amdgcn_wave_barrier();  // same-wave LDS RAW: keep program order

    // O += P @ V   (A = P[16 q][64 key], B = V dk-major)
#pragma unroll
    for (int ks = 0; ks < 2; ++ks) {
      const int aoff = ln * 128 + ks * 64 + hl * 16;
      v16bf af = ld_frag2(pw, aoff, aoff + 32);
#pragma unroll
      for (int t = 0; t < 4; ++t) {
        const int dkc  = t * 16 + ln;
        const int boff = dkc * 128 + ks * 64 + hl * 16;
        v16bf bf = ld_frag2(lds_v, boff, boff + 32);
        o[t] = wmma_bf16(af, bf, o[t]);
      }
    }
    // Next iteration's post-wait barrier separates lds_v / lds_k[buf] reuse.
  }

  // Epilogue: normalize, stage O tile in LDS (reuse lds_k[0]; the final loop
  // iteration only reads lds_k[1]/lds_v), then async-store LDS -> global.
  unsigned short* ost = &lds_k[0][0];   // [q][dk] 64x64 bf16
  float invl[8];
#pragma unroll
  for (int r = 0; r < 8; ++r) invl[r] = 1.0f / l[r];
#pragma unroll
  for (int t = 0; t < 4; ++t) {
    const int dkc = t * 16 + ln;
#pragma unroll
    for (int r = 0; r < 8; ++r)
      ost[(wave * 16 + hl * 8 + r) * 64 + dkc] = f2bf(o[t][r] * invl[r]);
  }
  __syncthreads();
  const unsigned long long so =
      (unsigned long long)out + (unsigned long long)rowbase * 1024ull * 2ull;
  const unsigned ob = lds_off(ost);
#pragma unroll
  for (int i = 0; i < 4; ++i) {
    int c   = i * 128 + tid;
    int row = c >> 3, cc = c & 7;
    unsigned lo = ob + row * 128 + cc * 16;
    unsigned vo = (unsigned)(((qt * 64 + row) * 1024 + hc + cc * 8) * 2);
    asm volatile("global_store_async_from_lds_b128 %0, %1, %2"
                 :: "v"(vo), "v"(lo), "s"(so) : "memory");
  }
  // S_ENDPGM performs an implicit wait-idle, draining ASYNCcnt.
}

// ---------------------------------------------------------------------------
// Launch: x[4096,1024] -> qkv[4096,3072](bf16 ws) -> attn[4096,1024](bf16 ws)
//         -> out[4096,1024] fp32.  Workspace: 32 MB.
// ---------------------------------------------------------------------------
extern "C" void kernel_launch(void* const* d_in, const int* in_sizes, int n_in,
                              void* d_out, int out_size, void* d_ws, size_t ws_size,
                              hipStream_t stream) {
  const float* x     = (const float*)d_in[0];
  const float* W_qkv = (const float*)d_in[1];
  const float* b_qkv = (const float*)d_in[2];
  const float* W_out = (const float*)d_in[3];
  const float* b_out = (const float*)d_in[4];
  float* out = (float*)d_out;

  unsigned short* qkv_ws  = (unsigned short*)d_ws;                 // [4096,3072]
  unsigned short* attn_ws = qkv_ws + (size_t)4096 * 3072;          // [4096,1024]

  dim3 blk(128);
  gemm_bias_kernel<float, unsigned short>
      <<<dim3(3072 / 64, 4096 / 64), blk, 0, stream>>>(x, W_qkv, b_qkv, qkv_ws,
                                                       4096, 3072, 1024);
  attn_kernel<<<dim3(32, 32), blk, 0, stream>>>(qkv_ws, attn_ws);
  gemm_bias_kernel<unsigned short, float>
      <<<dim3(1024 / 64, 4096 / 64), blk, 0, stream>>>(attn_ws, W_out, b_out,
                                                       out, 4096, 1024, 1024);
}
